// MambaIntegerBlock_48885317763160
// MI455X (gfx1250) — 8-bit matrix operand order repaired and checked by reading, compiled, not run on hardware
//
#include <hip/hip_runtime.h>
#include <stdint.h>

#define D_MODEL 768
#define D_STATE 16
#define DT_RANK 48
#define D_INNER 1536
#define ROWS    2048
#define XZ_COLS 3072

typedef int v8i __attribute__((ext_vector_type(8)));
typedef int v4i __attribute__((ext_vector_type(4)));

__device__ __forceinline__ int8_t quant8(float v) {
    v = fminf(fmaxf(v, -128.f), 127.f);
    return (int8_t)(int)rintf(v);
}

__device__ __forceinline__ void async_load_lds_b32(void* lds_dst, const void* gsrc) {
    unsigned ldsoff = (unsigned)(unsigned long long)(uintptr_t)lds_dst;
    unsigned long long ga = (unsigned long long)(uintptr_t)gsrc;
    asm volatile("global_load_async_to_lds_b32 %0, %1, off"
                 :: "v"(ldsoff), "v"(ga) : "memory");
}
__device__ __forceinline__ void async_store_lds_b32(void* gdst, const void* lds_src) {
    unsigned ldsoff = (unsigned)(unsigned long long)(uintptr_t)lds_src;
    unsigned long long ga = (unsigned long long)(uintptr_t)gdst;
    asm volatile("global_store_async_from_lds_b32 %0, %1, off"
                 :: "v"(ga), "v"(ldsoff) : "memory");
}
__device__ __forceinline__ void wait_asynccnt0() {
    asm volatile("s_wait_asynccnt 0x0" ::: "memory");
}

__global__ void zero_f32_kernel(float* p, int n) {
    int i = blockIdx.x * blockDim.x + threadIdx.x;
    if (i < n) p[i] = 0.f;
}

__global__ void abs_sum_kernel(const float* __restrict__ w, int n, float* __restrict__ out) {
    __shared__ float sbuf[256];
    float s = 0.f;
    for (int i = blockIdx.x * blockDim.x + threadIdx.x; i < n; i += gridDim.x * blockDim.x)
        s += fabsf(w[i]);
    sbuf[threadIdx.x] = s; __syncthreads();
    for (int off = 128; off > 0; off >>= 1) {
        if (threadIdx.x < off) sbuf[threadIdx.x] += sbuf[threadIdx.x + off];
        __syncthreads();
    }
    if (threadIdx.x == 0) atomicAdd(out, sbuf[0]);
}

__global__ void quant_weight_kernel(const float* __restrict__ w, int rows, int K, int Kpad,
                                    const float* __restrict__ sumSlot, float invN,
                                    float* __restrict__ swSlot, int8_t* __restrict__ q) {
    float sw = (*sumSlot) * invN + 1e-6f;
    if (blockIdx.x == 0 && threadIdx.x == 0) *swSlot = sw;
    float inv = 1.f / sw;
    int total = rows * Kpad;
    for (int i = blockIdx.x * blockDim.x + threadIdx.x; i < total; i += gridDim.x * blockDim.x) {
        int r = i / Kpad, c = i - r * Kpad;
        float v = (c < K) ? w[(size_t)r * K + c] * inv : 0.f;
        v = fminf(fmaxf(v, -1.f), 1.f);
        q[i] = (int8_t)(int)rintf(v);
    }
}

__global__ void norm_quant_kernel(const float* __restrict__ x, const float* __restrict__ gamma,
                                  int8_t* __restrict__ q, float* __restrict__ sx) {
    __shared__ float sbuf[256];
    int row = blockIdx.x;
    const float* xr = x + (size_t)row * D_MODEL;
    float v0 = xr[threadIdx.x], v1 = xr[threadIdx.x + 256], v2 = xr[threadIdx.x + 512];

    sbuf[threadIdx.x] = v0 + v1 + v2; __syncthreads();
    for (int off = 128; off > 0; off >>= 1) {
        if (threadIdx.x < off) sbuf[threadIdx.x] += sbuf[threadIdx.x + off];
        __syncthreads();
    }
    float mean = sbuf[0] * (1.f / D_MODEL); __syncthreads();

    float c0 = v0 - mean, c1 = v1 - mean, c2 = v2 - mean;
    sbuf[threadIdx.x] = c0 * c0 + c1 * c1 + c2 * c2; __syncthreads();
    for (int off = 128; off > 0; off >>= 1) {
        if (threadIdx.x < off) sbuf[threadIdx.x] += sbuf[threadIdx.x + off];
        __syncthreads();
    }
    float var = sbuf[0] * (1.f / D_MODEL); __syncthreads();

    float rms   = sqrtf(var + 1e-6f);
    float kk    = fmaxf(rintf(log2f(rms)), 0.f);
    float invr  = exp2f(-kk);
    float y0 = c0 * invr * gamma[threadIdx.x];
    float y1 = c1 * invr * gamma[threadIdx.x + 256];
    float y2 = c2 * invr * gamma[threadIdx.x + 512];

    sbuf[threadIdx.x] = fmaxf(fabsf(y0), fmaxf(fabsf(y1), fabsf(y2))); __syncthreads();
    for (int off = 128; off > 0; off >>= 1) {
        if (threadIdx.x < off) sbuf[threadIdx.x] = fmaxf(sbuf[threadIdx.x], sbuf[threadIdx.x + off]);
        __syncthreads();
    }
    float s_x = 127.f / (sbuf[0] + 1e-6f);
    if (threadIdx.x == 0) sx[row] = s_x;

    int8_t* qr = q + (size_t)row * D_MODEL;
    qr[threadIdx.x]       = quant8(y0 * s_x);
    qr[threadIdx.x + 256] = quant8(y1 * s_x);
    qr[threadIdx.x + 512] = quant8(y2 * s_x);
}

__global__ void row_quant_kernel(const float* __restrict__ in, int inStride, int colOff,
                                 int K, int Kpad, int8_t* __restrict__ q, float* __restrict__ sx) {
    __shared__ float sbuf[256];
    int row = blockIdx.x;
    const float* xr = in + (size_t)row * inStride + colOff;
    float am = 0.f;
    for (int c = threadIdx.x; c < K; c += blockDim.x) am = fmaxf(am, fabsf(xr[c]));
    sbuf[threadIdx.x] = am; __syncthreads();
    for (int off = blockDim.x >> 1; off > 0; off >>= 1) {
        if (threadIdx.x < off) sbuf[threadIdx.x] = fmaxf(sbuf[threadIdx.x], sbuf[threadIdx.x + off]);
        __syncthreads();
    }
    float s_x = 127.f / (sbuf[0] + 1e-6f);
    if (threadIdx.x == 0) sx[row] = s_x;
    int8_t* qr = q + (size_t)row * Kpad;
    for (int c = threadIdx.x; c < Kpad; c += blockDim.x)
        qr[c] = (c < K) ? quant8(xr[c] * s_x) : (int8_t)0;
}

template <int NT>
__global__ void gemm_iu8_kernel(const int8_t* __restrict__ A, const float* __restrict__ sxA,
                                const int8_t* __restrict__ W, const float* __restrict__ swSlot,
                                float* __restrict__ C, int M, int N, int K,
                                int ngroups, int colTileBase,
                                const float* __restrict__ residual,
                                const float* __restrict__ resGate) {
    int wave = (blockIdx.x * blockDim.x + threadIdx.x) >> 5;
    int lane = threadIdx.x & 31;
    int tm   = wave / ngroups;
    int tg   = wave - tm * ngroups;
    if (tm >= (M >> 4)) return;
    int tile0 = colTileBase + tg * NT;

    int laneLo = lane & 15;
    int hi     = lane >> 4;

    const int8_t* ap  = A + (size_t)(tm * 16 + laneLo) * K + hi * 8;
    const int8_t* bp0 = W + (size_t)(tile0 * 16 + laneLo) * K + hi * 8;

    v8i acc[NT];
#pragma unroll
    for (int s = 0; s < NT; ++s) acc[s] = (v8i){0, 0, 0, 0, 0, 0, 0, 0};

    for (int k = 0; k < K; k += 64) {
        v8i a;
#pragma unroll
        for (int p = 0; p < 4; ++p) {
            int2 r = *reinterpret_cast<const int2*>(ap + k + 16 * p);
            a[2 * p]     = r.x;
            a[2 * p + 1] = r.y;
        }
        __builtin_prefetch(ap + k + 64, 0, 1);
#pragma unroll
        for (int s = 0; s < NT; ++s) {
            const int8_t* bp = bp0 + (size_t)s * 16 * K + k;
            const v4i* pb = reinterpret_cast<const v4i*>(bp);
            (void)pb;
            v8i b;
#pragma unroll
            for (int p = 0; p < 4; ++p) {
                int2 r = *reinterpret_cast<const int2*>(bp + 16 * p);
                b[2 * p]     = r.x;
                b[2 * p + 1] = r.y;
            }
            __builtin_prefetch(bp + 64, 0, 1);
            acc[s] = __builtin_amdgcn_wmma_i32_16x16x64_iu8(true, a, true, b, acc[s],
                                                            false, false);
        }
    }

    float sw = *swSlot;
    float rg = resGate ? *resGate : 1.f;
#pragma unroll
    for (int v = 0; v < 8; ++v) {
        int m = tm * 16 + v + 8 * hi;
        float rowScale = sw / sxA[m];
#pragma unroll
        for (int s = 0; s < NT; ++s) {
            int n = (tile0 + s) * 16 + laneLo;
            float val = (float)acc[s][v] * rowScale;
            size_t idx = (size_t)m * N + n;
            if (residual) val = residual[idx] + val * rg;
            C[idx] = val;
        }
    }
}

__global__ void conv_act_quant_kernel(const float* __restrict__ xz, const float* __restrict__ cw,
                                      const float* __restrict__ cb, float* __restrict__ xact,
                                      int8_t* __restrict__ q, float* __restrict__ sx) {
    __shared__ float sbuf[256];
    int row = blockIdx.x;
    int b = row >> 10;
    int t = row & 1023;
    float vals[6];
    float am = 0.f;
#pragma unroll
    for (int e = 0; e < 6; ++e) {
        int d = threadIdx.x + e * 256;
        float acc = cb[d];
#pragma unroll
        for (int kk = 0; kk < 4; ++kk) {
            int tt = t - 3 + kk;
            if (tt >= 0)
                acc += xz[(size_t)(b * 1024 + tt) * XZ_COLS + d] * cw[d * 4 + kk];
        }
        acc = fminf(fmaxf(acc, -50.f), 50.f);
        acc = 0.5f * (acc + sqrtf(acc * acc + 4.f));
        vals[e] = acc;
        am = fmaxf(am, fabsf(acc));
    }
    sbuf[threadIdx.x] = am; __syncthreads();
    for (int off = 128; off > 0; off >>= 1) {
        if (threadIdx.x < off) sbuf[threadIdx.x] = fmaxf(sbuf[threadIdx.x], sbuf[threadIdx.x + off]);
        __syncthreads();
    }
    float s_x = 127.f / (sbuf[0] + 1e-6f);
    if (threadIdx.x == 0) sx[row] = s_x;
#pragma unroll
    for (int e = 0; e < 6; ++e) {
        int d = threadIdx.x + e * 256;
        xact[(size_t)row * D_INNER + d] = vals[e];
        q[(size_t)row * D_INNER + d]    = quant8(vals[e] * s_x);
    }
}

__global__ void scan_kernel(const float* __restrict__ xact, const float* __restrict__ dmod,
                            const float* __restrict__ proj, const float* __restrict__ xz,
                            const float* __restrict__ baseDecay, const float* __restrict__ shifts,
                            float* __restrict__ y) {
    __shared__ float xs[64][16], dm[64][16], Bs[64][16], Cs[64][16], zs[64][16], ys[64][16];
    int blk = blockIdx.x;
    int b   = blk / 96;
    int d0  = (blk % 96) * 16;
    int n   = threadIdx.x & 15;
    int dl  = (threadIdx.x >> 4) & 15;
    int d   = d0 + dl;

    float base = baseDecay[d * D_STATE + n];
    float pw   = exp2f(-shifts[d * D_STATE + n]);
    float h    = 0.f;

    for (int t0 = 0; t0 < 1024; t0 += 64) {
#pragma unroll
        for (int r = 0; r < 4; ++r) {
            int idx = threadIdx.x + r * 256;
            int i = idx >> 4, j = idx & 15;
            size_t rowg = (size_t)(b * 1024 + t0 + i);
            async_load_lds_b32(&xs[i][j], xact + rowg * D_INNER + d0 + j);
            async_load_lds_b32(&dm[i][j], dmod + rowg * D_INNER + d0 + j);
            async_load_lds_b32(&Bs[i][j], proj + rowg * 80 + DT_RANK + j);
            async_load_lds_b32(&Cs[i][j], proj + rowg * 80 + DT_RANK + D_STATE + j);
            async_load_lds_b32(&zs[i][j], xz + rowg * XZ_COLS + D_INNER + d0 + j);
        }
        wait_asynccnt0();
        __syncthreads();

        for (int i = 0; i < 64; ++i) {
            float dmv    = dm[i][dl];
            float dt_val = fminf(fmaxf(0.5f * (dmv + sqrtf(dmv * dmv + 4.f)) * 0.01f, 0.f), 0.1f);
            float dn     = fminf(fmaxf(base + dmv, 0.f), 32000.f);
            float decay  = dn * pw;
            float u      = fminf(fmaxf(xs[i][dl] * dt_val * Bs[i][n], -100.f), 100.f);
            h = decay * h + u;
            float hc   = fminf(fmaxf(h, -1000.f), 1000.f);
            float part = hc * Cs[i][n];
            part += __shfl_xor(part, 1, 32);
            part += __shfl_xor(part, 2, 32);
            part += __shfl_xor(part, 4, 32);
            part += __shfl_xor(part, 8, 32);
            if (n == 0) {
                float z    = zs[i][dl];
                float gate = 0.5f * (z * rsqrtf(z * z + 1.f) + 1.f);
                ys[i][dl]  = part * gate;
            }
        }
        __syncthreads();

#pragma unroll
        for (int r = 0; r < 4; ++r) {
            int idx = threadIdx.x + r * 256;
            int i = idx >> 4, j = idx & 15;
            async_store_lds_b32(y + (size_t)(b * 1024 + t0 + i) * D_INNER + d0 + j, &ys[i][j]);
        }
        wait_asynccnt0();
        __syncthreads();
    }
}

extern "C" void kernel_launch(void* const* d_in, const int* in_sizes, int n_in,
                              void* d_out, int out_size, void* d_ws, size_t ws_size,
                              hipStream_t stream) {
    (void)in_sizes; (void)n_in; (void)out_size; (void)ws_size;

    const float* hidden   = (const float*)d_in[0];
    const float* gamma    = (const float*)d_in[1];
    const float* in_w     = (const float*)d_in[2];
    const float* conv_w   = (const float*)d_in[3];
    const float* conv_b   = (const float*)d_in[4];
    const float* xproj_w  = (const float*)d_in[5];
    const float* dtproj_w = (const float*)d_in[6];
    const float* out_w    = (const float*)d_in[7];
    const float* baseDec  = (const float*)d_in[8];
    const float* resGate  = (const float*)d_in[9];
    const float* shifts   = (const float*)d_in[10];
    float* out = (float*)d_out;

    char* ws = (char*)d_ws;
    size_t off = 0;
    auto alloc = [&](size_t bytes) -> void* {
        off = (off + 255) & ~((size_t)255);
        void* p = ws + off;
        off += bytes;
        return p;
    };

    float*  red  = (float*)alloc(8 * sizeof(float));
    int8_t* qin  = (int8_t*)alloc((size_t)XZ_COLS * D_MODEL);
    int8_t* qx   = (int8_t*)alloc((size_t)80 * D_INNER);
    int8_t* qdt  = (int8_t*)alloc((size_t)D_INNER * 64);
    int8_t* qout = (int8_t*)alloc((size_t)D_MODEL * D_INNER);
    int8_t* hsq  = (int8_t*)alloc((size_t)ROWS * D_MODEL);
    float*  sx1  = (float*)alloc(ROWS * sizeof(float));
    float*  xz   = (float*)alloc((size_t)ROWS * XZ_COLS * sizeof(float));
    float*  xact = (float*)alloc((size_t)ROWS * D_INNER * sizeof(float));
    int8_t* xq2  = (int8_t*)alloc((size_t)ROWS * D_INNER);
    float*  sx2  = (float*)alloc(ROWS * sizeof(float));
    float*  proj = (float*)alloc((size_t)ROWS * 80 * sizeof(float));
    int8_t* dtq  = (int8_t*)alloc((size_t)ROWS * 64);
    float*  sx3  = (float*)alloc(ROWS * sizeof(float));
    float*  dmod = (float*)alloc((size_t)ROWS * D_INNER * sizeof(float));
    float*  ybuf = (float*)alloc((size_t)ROWS * D_INNER * sizeof(float));
    int8_t* yq   = (int8_t*)alloc((size_t)ROWS * D_INNER);
    float*  sx4  = (float*)alloc(ROWS * sizeof(float));

    zero_f32_kernel<<<1, 32, 0, stream>>>(red, 8);
    abs_sum_kernel<<<512, 256, 0, stream>>>(in_w,     XZ_COLS * D_MODEL,  red + 0);
    abs_sum_kernel<<<128, 256, 0, stream>>>(xproj_w,  80 * D_INNER,       red + 1);
    abs_sum_kernel<<<128, 256, 0, stream>>>(dtproj_w, D_INNER * DT_RANK,  red + 2);
    abs_sum_kernel<<<256, 256, 0, stream>>>(out_w,    D_MODEL * D_INNER,  red + 3);

    quant_weight_kernel<<<1024, 256, 0, stream>>>(in_w, XZ_COLS, D_MODEL, D_MODEL,
                                                  red + 0, 1.f / (XZ_COLS * D_MODEL), red + 4, qin);
    quant_weight_kernel<<<128, 256, 0, stream>>>(xproj_w, 80, D_INNER, D_INNER,
                                                 red + 1, 1.f / (80 * D_INNER), red + 5, qx);
    quant_weight_kernel<<<128, 256, 0, stream>>>(dtproj_w, D_INNER, DT_RANK, 64,
                                                 red + 2, 1.f / (D_INNER * DT_RANK), red + 6, qdt);
    quant_weight_kernel<<<512, 256, 0, stream>>>(out_w, D_MODEL, D_INNER, D_INNER,
                                                 red + 3, 1.f / (D_MODEL * D_INNER), red + 7, qout);

    norm_quant_kernel<<<ROWS, 256, 0, stream>>>(hidden, gamma, hsq, sx1);

    gemm_iu8_kernel<4><<<768, 256, 0, stream>>>(hsq, sx1, qin, red + 4, xz,
                                                ROWS, XZ_COLS, D_MODEL, 48, 0,
                                                nullptr, nullptr);

    conv_act_quant_kernel<<<ROWS, 256, 0, stream>>>(xz, conv_w, conv_b, xact, xq2, sx2);

    gemm_iu8_kernel<4><<<16, 256, 0, stream>>>(xq2, sx2, qx, red + 5, proj,
                                               ROWS, 80, D_INNER, 1, 0,
                                               nullptr, nullptr);
    gemm_iu8_kernel<1><<<16, 256, 0, stream>>>(xq2, sx2, qx, red + 5, proj,
                                               ROWS, 80, D_INNER, 1, 4,
                                               nullptr, nullptr);

    row_quant_kernel<<<ROWS, 64, 0, stream>>>(proj, 80, 0, DT_RANK, 64, dtq, sx3);

    gemm_iu8_kernel<4><<<384, 256, 0, stream>>>(dtq, sx3, qdt, red + 6, dmod,
                                                ROWS, D_INNER, 64, 24, 0,
                                                nullptr, nullptr);

    scan_kernel<<<192, 256, 0, stream>>>(xact, dmod, proj, xz, baseDec, shifts, ybuf);

    row_quant_kernel<<<ROWS, 256, 0, stream>>>(ybuf, D_INNER, 0, D_INNER, D_INNER, yq, sx4);

    gemm_iu8_kernel<4><<<192, 256, 0, stream>>>(yq, sx4, qout, red + 7, out,
                                                ROWS, D_MODEL, D_INNER, 12, 0,
                                                hidden, resGate);
}
